// ContextPosSelfAttn_17678085390588
// MI455X (gfx1250) — matrix operand order repaired and checked by reading, not compiled, not run on hardware
//
#include <hip/hip_runtime.h>
#include <hip/hip_bf16.h>

typedef __attribute__((ext_vector_type(16))) __bf16 v16bf;
typedef __attribute__((ext_vector_type(8)))  float  v8f;

#define LQ   1024
#define DH   64
#define NPOS 1025
#define PSTR 1032   // pos-table LDS stride (1032 % 64 = 8 -> conflict-free rows)
#define PPAD 20     // P-buffer stride: multiple of 4 floats -> 16B-aligned rows for b128

static __device__ __forceinline__ v8f wmma_bf16(v16bf a, v16bf b, v8f c) {
  return __builtin_amdgcn_wmma_f32_16x16x32_bf16(
      /*neg_a=*/false, a, /*neg_b=*/false, b,
      /*c_mod=*/(short)0, c, /*reuse_a=*/false, /*reuse_b=*/false);
}

__global__ __launch_bounds__(32)
void cope_attn_kernel(const float* __restrict__ Q,
                      const float* __restrict__ K,
                      const float* __restrict__ Kc,
                      const float* __restrict__ V,
                      const float* __restrict__ PE,
                      float* __restrict__ Out)
{
  extern __shared__ float smem[];
  float* postab   = smem;                    // 16 x PSTR (pos-logit table, f32)
  float* g_tile   = postab + 16 * PSTR;      // 16 x 17   (gates)
  float* pos_tile = g_tile + 16 * 17;        // 16 x 17   (positions)
  float* p_buf    = pos_tile + 16 * 17;      // 2 x 16 x PPAD (double-buffered P)
  float* alpha_s  = p_buf + 2 * 16 * PPAD;   // 16        (per-row rescale)

  const int lane = threadIdx.x & 31;
  const int hf   = lane >> 4;   // half-wave: 0 or 1
  const int l16  = lane & 15;
  const int qt   = blockIdx.x;  // query tile index (16 rows)
  const int b    = blockIdx.y;  // batch*head
  const int q0   = qt * 16;
  const float scale = 0.125f;   // 1/sqrt(64)

  const float* Qb  = Q  + (size_t)b * LQ * DH;
  const float* Kb  = K  + (size_t)b * LQ * DH;
  const float* Kcb = Kc + (size_t)b * LQ * DH;
  const float* Vb  = V  + (size_t)b * LQ * DH;

  // ---- Q tile as two bf16 A-fragments (K chunks 0..31 and 32..63) ----
  // A k order used here (B matches; not the ISA table order): m = lane&15, elem e: pair p=e>>1,
  // k = hf*16 + p*2 + (e&1) -> element pairs contiguous.
  v16bf a0, a1;
  {
    const float* qrow = Qb + (size_t)(q0 + l16) * DH;
    #pragma unroll
    for (int p = 0; p < 8; ++p) {
      int k = hf * 16 + p * 2;
      float2 x0 = *(const float2*)(qrow + k);
      float2 x1 = *(const float2*)(qrow + 32 + k);
      a0[2*p]   = (__bf16)x0.x;  a0[2*p+1] = (__bf16)x0.y;
      a1[2*p]   = (__bf16)x1.x;  a1[2*p+1] = (__bf16)x1.y;
    }
  }

  // ---- Phase 1: pos table = Q_tile @ pos_emb (16 x 1025) into LDS ----
  for (int nt = 0; nt < 64; ++nt) {
    int col = nt * 16 + l16;
    v16bf b0, b1;
    #pragma unroll
    for (int e = 0; e < 16; ++e) {
      int k = hf * 16 + e;                        // B: same k order as A, k = hf*16+e, n = l16
      b0[e] = (__bf16)PE[(size_t)k        * NPOS + col];
      b1[e] = (__bf16)PE[(size_t)(k + 32) * NPOS + col];
    }
    v8f c = {};
    c = wmma_bf16(a0, b0, c);
    c = wmma_bf16(a1, b1, c);
    #pragma unroll
    for (int r = 0; r < 8; ++r)                    // C layout: m = hf*8+r, n = l16
      postab[(hf * 8 + r) * PSTR + col] = c[r];
  }
  // tail column p = 1024 as a plain dot product (16 rows, lanes 0..15).
  if (lane < 16) {
    const float* qrow = Qb + (size_t)(q0 + lane) * DH;
    float acc = 0.f;
    #pragma unroll
    for (int k = 0; k < DH; ++k)
      acc += qrow[k] * PE[(size_t)k * NPOS + (NPOS - 1)];
    postab[lane * PSTR + (NPOS - 1)] = acc;
  }
  __syncthreads();

  // ---- Phase 2: key tiles right-to-left; flash accumulation + suffix carry ----
  float carry = 0.f;          // per-row gate suffix sum (threads 0..15 own row=lane)
  float mrow[8], rsum[8];
  v8f o[4];
  #pragma unroll
  for (int r = 0; r < 8; ++r) { mrow[r] = -1e30f; rsum[r] = 0.f; }
  #pragma unroll
  for (int nc = 0; nc < 4; ++nc) o[nc] = (v8f){};

  bool pending = false;       // deferred P tile waiting for its fused K=32 P@V
  int  c0_pend = 0;

  for (int kt = qt; kt >= 0; --kt) {
    int c0 = kt * 16;
    float* p_cur = p_buf + (kt & 1) * 16 * PPAD;

    // prefetch next tile's K/Kc rows (speculative, dropped if useless)
    if (kt > 0) {
      __builtin_prefetch(Kb  + (size_t)(c0 - 16 + l16) * DH + hf * 16, 0, 1);
      __builtin_prefetch(Kcb + (size_t)(c0 - 16 + l16) * DH + hf * 16, 0, 1);
    }

    // B-fragments for key_cope and key: B[k][n] = Kmat[c0+n][k];
    // per lane the 16 elements are contiguous in k -> four b128 per chunk.
    v16bf bc0, bc1, bk0, bk1;
    {
      const float* kcrow = Kcb + (size_t)(c0 + l16) * DH + hf * 16;
      const float* krow  = Kb  + (size_t)(c0 + l16) * DH + hf * 16;
      #pragma unroll
      for (int i = 0; i < 4; ++i) {
        float4 xc0 = *(const float4*)(kcrow + 4 * i);
        float4 xc1 = *(const float4*)(kcrow + 32 + 4 * i);
        float4 xk0 = *(const float4*)(krow  + 4 * i);
        float4 xk1 = *(const float4*)(krow  + 32 + 4 * i);
        bc0[4*i] = (__bf16)xc0.x; bc0[4*i+1] = (__bf16)xc0.y; bc0[4*i+2] = (__bf16)xc0.z; bc0[4*i+3] = (__bf16)xc0.w;
        bc1[4*i] = (__bf16)xc1.x; bc1[4*i+1] = (__bf16)xc1.y; bc1[4*i+2] = (__bf16)xc1.z; bc1[4*i+3] = (__bf16)xc1.w;
        bk0[4*i] = (__bf16)xk0.x; bk0[4*i+1] = (__bf16)xk0.y; bk0[4*i+2] = (__bf16)xk0.z; bk0[4*i+3] = (__bf16)xk0.w;
        bk1[4*i] = (__bf16)xk1.x; bk1[4*i+1] = (__bf16)xk1.y; bk1[4*i+2] = (__bf16)xk1.z; bk1[4*i+3] = (__bf16)xk1.w;
      }
    }

    v8f sg = {}, sk = {};
    sg = wmma_bf16(a0, bc0, sg);
    sg = wmma_bf16(a1, bc1, sg);
    sk = wmma_bf16(a0, bk0, sk);
    sk = wmma_bf16(a1, bk1, sk);

    // gates = sigmoid(gate_logits * scale) * causal_mask (mask via value select)
    #pragma unroll
    for (int r = 0; r < 8; ++r) {
      int m = hf * 8 + r;
      int l = q0 + m, c = c0 + l16;
      float gate = 1.f / (1.f + __expf(-sg[r] * scale));
      g_tile[m * 17 + l16] = (c <= l) ? gate : 0.f;
    }
    __syncthreads();

    // per-row suffix scan within tile (+ carry from tiles to the right)
    if (lane < 16) {
      float run = 0.f;
      #pragma unroll
      for (int j = 15; j >= 0; --j) {
        run += g_tile[lane * 17 + j];
        pos_tile[lane * 17 + j] = fminf(carry + run, (float)(NPOS - 1));
      }
      carry += run;
    }
    __syncthreads();

    // pos-logit lerp + masked scaled logits
    float pr[8];
    #pragma unroll
    for (int r = 0; r < 8; ++r) {
      int m = hf * 8 + r;
      int l = q0 + m, c = c0 + l16;
      float pos = pos_tile[m * 17 + l16];
      float pf  = floorf(pos);
      int   i0  = (int)pf;
      int   i1  = (int)ceilf(pos);
      float w   = pos - pf;
      float lg  = postab[m * PSTR + i1] * w + postab[m * PSTR + i0] * (1.f - w);
      float s   = (sk[r] + lg) * scale;
      pr[r] = (c <= l) ? s : -1e30f;
    }

    // online softmax: 16-lane reductions match the C-layout half-wave row split
    float al[8];
    #pragma unroll
    for (int r = 0; r < 8; ++r) {
      float v = pr[r];
      #pragma unroll
      for (int off = 1; off < 16; off <<= 1)
        v = fmaxf(v, __shfl_xor(v, off, 16));
      float mnew  = fmaxf(mrow[r], v);
      float alpha = __expf(mrow[r] - mnew);
      float p     = __expf(pr[r] - mnew);
      float ps = p;
      #pragma unroll
      for (int off = 1; off < 16; off <<= 1)
        ps += __shfl_xor(ps, off, 16);
      rsum[r] = rsum[r] * alpha + ps;
      mrow[r] = mnew;
      al[r]   = alpha;
      #pragma unroll
      for (int nc = 0; nc < 4; ++nc) o[nc][r] *= alpha;
      p_cur[(hf * 8 + r) * PPAD + l16] = p;
    }
    if (l16 == 0) {
      #pragma unroll
      for (int r = 0; r < 8; ++r) alpha_s[hf * 8 + r] = al[r];
    }
    __syncthreads();

    if (!pending) {
      pending = true;  c0_pend = c0;     // defer this tile's P@V
    } else {
      // Fused K=32 P@V over keys c0..c0+31:
      //   A k<16  -> P_cur (keys c0..c0+15), current scale;
      //   A k>=16 -> P_prev (keys c0+16..c0+31) rescaled by this iter's alpha.
      const float* p_prev = p_buf + ((kt + 1) & 1) * 16 * PPAD;
      float am = alpha_s[l16];
      float4 c0v = *(const float4*)&p_cur [l16 * PPAD + hf * 8];
      float4 c1v = *(const float4*)&p_cur [l16 * PPAD + hf * 8 + 4];
      float4 q0v = *(const float4*)&p_prev[l16 * PPAD + hf * 8];
      float4 q1v = *(const float4*)&p_prev[l16 * PPAD + hf * 8 + 4];
      v16bf ap;
      ap[0]  = (__bf16)c0v.x; ap[1]  = (__bf16)c0v.y; ap[2]  = (__bf16)c0v.z; ap[3]  = (__bf16)c0v.w;
      ap[4]  = (__bf16)c1v.x; ap[5]  = (__bf16)c1v.y; ap[6]  = (__bf16)c1v.z; ap[7]  = (__bf16)c1v.w;
      ap[8]  = (__bf16)(q0v.x * am); ap[9]  = (__bf16)(q0v.y * am);
      ap[10] = (__bf16)(q0v.z * am); ap[11] = (__bf16)(q0v.w * am);
      ap[12] = (__bf16)(q1v.x * am); ap[13] = (__bf16)(q1v.y * am);
      ap[14] = (__bf16)(q1v.z * am); ap[15] = (__bf16)(q1v.w * am);
      // V B-fragments: full K=32, every lane loads valid data (no selects).
      const float* vbase = Vb + (size_t)(c0 + hf * 8) * DH + l16; // row c0+hf*8+(e<8?e:e+8)
      #pragma unroll
      for (int nc = 0; nc < 4; ++nc) {
        v16bf bv;
        #pragma unroll
        for (int e = 0; e < 16; ++e)
          bv[e] = (__bf16)vbase[(size_t)(e < 8 ? e : e + 8) * DH + nc * 16];
        o[nc] = wmma_bf16(ap, bv, o[nc]);
      }
      pending = false;
    }
  }

  // flush a leftover deferred tile (happens iff qt is even; then c0_pend == 0)
  if (pending) {
    const float* pp = p_buf + ((c0_pend >> 4) & 1) * 16 * PPAD;
    float4 q0v = *(const float4*)&pp[l16 * PPAD + hf * 8];
    float4 q1v = *(const float4*)&pp[l16 * PPAD + hf * 8 + 4];
    v16bf ap;
    ap[0] = (__bf16)q0v.x; ap[1] = (__bf16)q0v.y; ap[2] = (__bf16)q0v.z; ap[3] = (__bf16)q0v.w;
    ap[4] = (__bf16)q1v.x; ap[5] = (__bf16)q1v.y; ap[6] = (__bf16)q1v.z; ap[7] = (__bf16)q1v.w;
    #pragma unroll
    for (int e = 8; e < 16; ++e) ap[e] = (__bf16)0.f;   // k >= 16 -> 0
    const float* vbase = Vb + (size_t)(c0_pend + hf * 8) * DH + l16;
    #pragma unroll
    for (int nc = 0; nc < 4; ++nc) {
      v16bf bv;
      #pragma unroll
      for (int e = 0; e < 16; ++e)
        bv[e] = (__bf16)vbase[(size_t)(e < 8 ? e : e + 8) * DH + nc * 16]; // rows < 32, always in-bounds
      o[nc] = wmma_bf16(ap, bv, o[nc]);
    }
  }

  // ---- epilogue: normalize and store ----
  #pragma unroll
  for (int r = 0; r < 8; ++r) {
    int m = hf * 8 + r;
    float inv = 1.f / rsum[r];
    #pragma unroll
    for (int nc = 0; nc < 4; ++nc) {
      Out[((size_t)b * LQ + q0 + m) * DH + nc * 16 + l16] = o[nc][r] * inv;
    }
  }
}

extern "C" void kernel_launch(void* const* d_in, const int* in_sizes, int n_in,
                              void* d_out, int out_size, void* d_ws, size_t ws_size,
                              hipStream_t stream) {
  const float* Q  = (const float*)d_in[0];
  const float* K  = (const float*)d_in[1];
  const float* Kc = (const float*)d_in[2];
  const float* V  = (const float*)d_in[3];
  const float* PE = (const float*)d_in[4];
  // d_in[5] (attn_mask) is causal-tril; mask applied analytically in-kernel.

  dim3 grid(LQ / 16, 64);   // (query tiles, batch*heads)
  dim3 block(32);           // one wave32 per workgroup
  size_t shmem = (size_t)(16 * PSTR + 2 * 16 * 17 + 2 * 16 * PPAD + 16) * sizeof(float);
  hipLaunchKernelGGL(cope_attn_kernel, grid, block, shmem, stream,
                     Q, K, Kc, V, PE, (float*)d_out);
}
